// ComfiFastGRNNTorch_53712861003941
// MI455X (gfx1250) — matrix operand order repaired and checked by reading, not compiled, not run on hardware
//
#include <hip/hip_runtime.h>
#include <hip/hip_bf16.h>
#include <math.h>

// Problem sizes (fixed by the reference)
#define B_DIM 64
#define S_DIM 2048
#define I_DIM 64
#define H_DIM 128

#define LOG2E 1.44269504088896340736f

typedef __attribute__((ext_vector_type(16))) __bf16         v16bf;
typedef __attribute__((ext_vector_type(8)))  float          v8f;
typedef __attribute__((ext_vector_type(8)))  unsigned short ushort8;

// float -> bf16 bits, round-to-nearest-even
static __device__ __forceinline__ unsigned short f2bf(float f) {
    unsigned int x = __float_as_uint(f);
    unsigned int r = x + 0x7FFFu + ((x >> 16) & 1u);
    return (unsigned short)(r >> 16);
}

union Frag16 {
    v16bf v;
    ushort8 h[2];
    unsigned short s[16];
};

// A-matrix fragment (16x32 bf16, wave32 layout) for one K-block of 32,
// loaded from a bf16 row stored contiguously in LDS.
// Lanes 0-15 hold K = {kb*32+0..7, kb*32+16..23}, lanes 16-31 hold
// K = {kb*32+8..15, kb*32+24..31}: two contiguous 16B runs -> 2x ds_load_b128.
static __device__ __forceinline__ v16bf lds_a_frag(const unsigned short* rowp,
                                                   int kb, int laneHi) {
    Frag16 u;
    u.h[0] = *(const ushort8*)(rowp + kb * 32 + 8 * laneHi);
    u.h[1] = *(const ushort8*)(rowp + kb * 32 + 16 + 8 * laneHi);
    return u.v;
}

// B-matrix fragment (32x16 bf16) for one K-block, from a row-major f32
// weight matrix W[K][ldn]; lane holds column ncol, K = kb*32 + 8*laneHi + e (e<8), kb*32 + 16 + 8*laneHi + (e-8) (e>=8).
static __device__ __forceinline__ v16bf glb_b_frag(const float* __restrict__ W,
                                                   int kb, int laneHi, int ncol,
                                                   int ldn) {
    Frag16 u;
    const int kbase = kb * 32 + 8 * laneHi;
#pragma unroll
    for (int e = 0; e < 16; ++e)
        u.s[e] = f2bf(W[(size_t)(kbase + e + (e >= 8 ? 8 : 0)) * ldn + ncol]);
    return u.v;
}

// Branch-free fast sigmoid: 1 / (1 + 2^(-x*log2e)) -- native v_exp/v_rcp.
static __device__ __forceinline__ float fast_sigmoid(float x) {
    return __builtin_amdgcn_rcpf(1.0f + __builtin_amdgcn_exp2f(-x * LOG2E));
}

// ---------------------------------------------------------------------------
// Projection GEMM: Y[R][128] = X[R][K] @ W[K][128], bf16 WMMA, f32 accum.
// One block = 16 rows (one M tile); 8 waves = 8 N tiles of 16.
// ---------------------------------------------------------------------------
template <int K>
__global__ __launch_bounds__(256) void proj_gemm_kernel(
    const float* __restrict__ X, const float* __restrict__ W,
    float* __restrict__ Y) {
    __shared__ unsigned short xbf[16 * K];

    const int tid = threadIdx.x;
    const int wave = tid >> 5, lane = tid & 31;
    const int laneHi = lane >> 4, lr = lane & 15;
    const long rowbase = (long)blockIdx.x * 16;

    // Stage the 16xK input tile into LDS as bf16 (row-major).
    for (int i = tid; i < 16 * K; i += 256) {
        const int r = i / K, k = i % K;
        xbf[i] = f2bf(X[(rowbase + r) * K + k]);
    }
    __syncthreads();

    const int ncol = wave * 16 + lr;
    v8f c = {};
#pragma unroll
    for (int kb = 0; kb < K / 32; ++kb) {
        v16bf a = lds_a_frag(xbf + lr * K, kb, laneHi);
        v16bf b = glb_b_frag(W, kb, laneHi, ncol, H_DIM);
        c = __builtin_amdgcn_wmma_f32_16x16x32_bf16(false, a, false, b,
                                                    (short)0, c, false, false);
    }
#pragma unroll
    for (int r = 0; r < 8; ++r) {
        const long gm = rowbase + r + 8 * laneHi;
        Y[gm * H_DIM + ncol] = c[r];
    }
}

// ---------------------------------------------------------------------------
// Sequential recurrence for one layer. One block = 16 batch rows, 8 waves
// cover all 128 hidden columns (one 16x16 N tile each).
//  - h bf16 double-buffered in LDS -> ONE barrier per step
//  - u held in per-wave B-fragments for the whole scan
//  - wx_t prefetched one step ahead (overlaps WMMA + gate math)
//  - branch-free native-exp gates (keeps EXEC all-1s as WMMA requires)
// ---------------------------------------------------------------------------
__global__ __launch_bounds__(256) void rnn_scan_kernel(
    const float* __restrict__ WX, const float* __restrict__ U,
    const float* __restrict__ bg, const float* __restrict__ bu,
    const float* __restrict__ zeta, const float* __restrict__ nu,
    const float* __restrict__ lambd, const float* __restrict__ gamma,
    float* __restrict__ OUT, float* __restrict__ HT) {
    __shared__ unsigned short hbf[2][16 * H_DIM];  // 2 x 4 KB bf16 hidden state

    const int tid = threadIdx.x;
    const int wave = tid >> 5, lane = tid & 31;
    const int laneHi = lane >> 4, lr = lane & 15;
    const int ncol = wave * 16 + lr;
    const long bbase = (long)blockIdx.x * 16;

    // Per-lane constants, pre-scaled for exp2-based gates.
    const float bgl = -bg[ncol] * LOG2E;          // exp2(pre*-L2E + bgl)
    const float bul = 2.0f * bu[ncol] * LOG2E;    // exp2(pre*2L2E + bul)
    const float sz = fast_sigmoid(zeta[0]);
    const float sn = fast_sigmoid(nu[0]);
    float gc = gamma[0];
    gc = fminf(fmaxf(gc, 0.0f), 1.0f);
    const float cl = (1.0f - gc) * lambd[0];
    const float szn = sz + sn;

    // Recurrent weight fragments: constant across all 2048 steps.
    v16bf bfrag[4];
#pragma unroll
    for (int kb = 0; kb < 4; ++kb)
        bfrag[kb] = glb_b_frag(U, kb, laneHi, ncol, H_DIM);

    // h = 0 (buffer 0 is read at t=0)
    for (int i = tid; i < 16 * H_DIM; i += 256) hbf[0][i] = 0;
    float hreg[8];
    long wxbase[8];
#pragma unroll
    for (int r = 0; r < 8; ++r) {
        hreg[r] = 0.0f;
        wxbase[r] = ((bbase + r + 8 * laneHi) * S_DIM) * H_DIM + ncol;
    }
    __syncthreads();

    // Prefetch wx for t=0.
    float cw[8];
#pragma unroll
    for (int r = 0; r < 8; ++r) cw[r] = WX[wxbase[r]];

    int cur = 0;
    for (int t = 0; t < S_DIM; ++t) {
        const int nxt = cur ^ 1;
        const unsigned short* hrow = hbf[cur] + lr * H_DIM;
        v16bf a0 = lds_a_frag(hrow, 0, laneHi);
        v16bf a1 = lds_a_frag(hrow, 1, laneHi);
        v16bf a2 = lds_a_frag(hrow, 2, laneHi);
        v16bf a3 = lds_a_frag(hrow, 3, laneHi);

        // c0 starts from wx_t; then prefetch wx_{t+1} (overlaps the WMMAs).
        v8f c0, c1 = {};
#pragma unroll
        for (int r = 0; r < 8; ++r) c0[r] = cw[r];
        const int tn = (t + 1 < S_DIM) ? (t + 1) : t;
#pragma unroll
        for (int r = 0; r < 8; ++r) cw[r] = WX[wxbase[r] + (long)tn * H_DIM];

        // Two independent 2-deep WMMA chains (shorter dependency chain).
        c0 = __builtin_amdgcn_wmma_f32_16x16x32_bf16(false, a0, false, bfrag[0],
                                                     (short)0, c0, false, false);
        c1 = __builtin_amdgcn_wmma_f32_16x16x32_bf16(false, a2, false, bfrag[2],
                                                     (short)0, c1, false, false);
        c0 = __builtin_amdgcn_wmma_f32_16x16x32_bf16(false, a1, false, bfrag[1],
                                                     (short)0, c0, false, false);
        c1 = __builtin_amdgcn_wmma_f32_16x16x32_bf16(false, a3, false, bfrag[3],
                                                     (short)0, c1, false, false);

        // Gates + state update, fully branch-free.
#pragma unroll
        for (int r = 0; r < 8; ++r) {
            const int m = r + 8 * laneHi;
            const float pre = c0[r] + c1[r];
            // z = sigmoid(pre + bg)
            const float ez = __builtin_amdgcn_exp2f(fmaf(pre, -LOG2E, bgl));
            const float z = __builtin_amdgcn_rcpf(1.0f + ez);
            // hh = tanh(pre + bu) = 1 - 2/(exp(2(pre+bu)) + 1)
            const float et = __builtin_amdgcn_exp2f(fmaf(pre, 2.0f * LOG2E, bul));
            const float hh = 1.0f - 2.0f * __builtin_amdgcn_rcpf(1.0f + et);
            // h_new = z*h + (sz*(1-z)+sn)*hh ; coef = szn - sz*z
            const float coef = fmaf(-sz, z, szn);
            const float hn = fmaf(z, hreg[r], coef * hh);
            const float hc = fmaf(gc, hn, cl);
            hreg[r] = hc;
            hbf[nxt][m * H_DIM + ncol] = f2bf(hc);
            OUT[wxbase[r] + (long)t * H_DIM] = hc;
        }
        __syncthreads();  // writes to hbf[nxt] visible; iter-t reads all done
        cur = nxt;
    }

    // Final hidden state h_T
#pragma unroll
    for (int r = 0; r < 8; ++r) {
        const int m = r + 8 * laneHi;
        HT[(bbase + m) * H_DIM + ncol] = hreg[r];
    }
}

// ---------------------------------------------------------------------------
extern "C" void kernel_launch(void* const* d_in, const int* in_sizes, int n_in,
                              void* d_out, int out_size, void* d_ws,
                              size_t ws_size, hipStream_t stream) {
    const float* x   = (const float*)d_in[0];
    const float* w0  = (const float*)d_in[1];
    const float* u0  = (const float*)d_in[2];
    const float* bg0 = (const float*)d_in[3];
    const float* bu0 = (const float*)d_in[4];
    const float* z0  = (const float*)d_in[5];
    const float* n0  = (const float*)d_in[6];
    const float* l0  = (const float*)d_in[7];
    const float* g0  = (const float*)d_in[8];
    const float* w1  = (const float*)d_in[9];
    const float* u1  = (const float*)d_in[10];
    const float* bg1 = (const float*)d_in[11];
    const float* bu1 = (const float*)d_in[12];
    const float* z1  = (const float*)d_in[13];
    const float* n1  = (const float*)d_in[14];
    const float* l1  = (const float*)d_in[15];
    const float* g1  = (const float*)d_in[16];

    float* out = (float*)d_out;                       // [B,S,H]; out0 then out1
    float* hT  = out + (long)B_DIM * S_DIM * H_DIM;   // [2,B,H]
    float* wx  = (float*)d_ws;                        // [B,S,H] f32 scratch

    const long R = (long)B_DIM * S_DIM;
    dim3 blk(256);

    // Layer 0
    proj_gemm_kernel<I_DIM><<<dim3((unsigned)(R / 16)), blk, 0, stream>>>(x, w0, wx);
    rnn_scan_kernel<<<dim3(B_DIM / 16), blk, 0, stream>>>(wx, u0, bg0, bu0, z0,
                                                          n0, l0, g0, out, hT);
    // Layer 1 (reads out0 from d_out, then overwrites d_out with out1)
    proj_gemm_kernel<H_DIM><<<dim3((unsigned)(R / 16)), blk, 0, stream>>>(out, w1, wx);
    rnn_scan_kernel<<<dim3(B_DIM / 16), blk, 0, stream>>>(
        wx, u1, bg1, bu1, z1, n1, l1, g1, out, hT + (long)B_DIM * H_DIM);
}
